// TTTWithAdaptiveNorm_70291434766516
// MI455X (gfx1250) — compile-verified, matrix operand order checked by reading, not run on hardware
//
#include <hip/hip_runtime.h>

#define Bsz 2
#define Lseq 2048
#define HID 2048
#define NH 16
#define HD 128
#define CD 256
#define EPSF 1e-6f

typedef __attribute__((ext_vector_type(16))) __bf16 v16bf;
typedef __attribute__((ext_vector_type(8)))  float  v8f;
typedef __attribute__((ext_vector_type(4)))  int    v4i;

union Frag { v16bf v; unsigned int u[8]; };

__device__ __forceinline__ unsigned short f2bf(float f) {
  unsigned int u = __float_as_uint(f);
  u += 0x7FFFu + ((u >> 16) & 1u);   // round-to-nearest-even
  return (unsigned short)(u >> 16);
}

// ---- CDNA5 async global->LDS copy (ASYNCcnt path) -------------------------
__device__ __forceinline__ void async_b128(unsigned short* lds, const unsigned short* g) {
#if __has_builtin(__builtin_amdgcn_global_load_async_to_lds_b128)
  __builtin_amdgcn_global_load_async_to_lds_b128(
      (__attribute__((address_space(1))) v4i*)(size_t)g,
      (__attribute__((address_space(3))) v4i*)(unsigned int)(size_t)lds,
      0, 0);
#else
  // VDST = LDS byte address (low 32 bits of generic LDS pointer), VADDR = 64-bit global addr
  asm volatile("global_load_async_to_lds_b128 %0, %1, off"
               :: "v"((unsigned int)(size_t)lds),
                  "v"((unsigned long long)(size_t)g)
               : "memory");
#endif
}

template <int N>
__device__ __forceinline__ void wait_asynccnt() {
#if __has_builtin(__builtin_amdgcn_s_wait_asynccnt)
  __builtin_amdgcn_s_wait_asynccnt(N);
#else
  asm volatile("s_wait_asynccnt %0" :: "i"(N) : "memory");
#endif
}
// ---------------------------------------------------------------------------

__device__ __forceinline__ float blkSum(float v, float* s) {
  int t = threadIdx.x;
  s[t] = v; __syncthreads();
  for (int off = blockDim.x >> 1; off > 0; off >>= 1) {
    if (t < off) s[t] += s[t + off];
    __syncthreads();
  }
  float r = s[0]; __syncthreads();
  return r;
}

// mod = cond @ dense_w.T + dense_b ; split into scale/shift (ws) and gate (d_out tail)
__global__ void mod_kernel(const float* __restrict__ cond, const float* __restrict__ dw,
                           const float* __restrict__ db, float* __restrict__ scale,
                           float* __restrict__ shift, float* __restrict__ gate_out) {
  int o = blockIdx.x * blockDim.x + threadIdx.x;
  if (o >= Bsz * 3 * HID) return;
  int b = o / (3 * HID);
  int j = o % (3 * HID);
  const float* c = cond + b * CD;
  const float* w = dw + (size_t)j * CD;
  float acc = db[j];
  for (int k = 0; k < CD; ++k) acc += c[k] * w[k];
  if (j < HID)            scale[b * HID + j] = acc;
  else if (j < 2 * HID)   shift[b * HID + (j - HID)] = acc;
  else                    gate_out[b * HID + (j - 2 * HID)] = acc;
}

// adaptive RMSNorm + modulation -> bf16 x
__global__ __launch_bounds__(256) void adanorm_kernel(const float* __restrict__ hs,
    const float* __restrict__ scale, const float* __restrict__ shift,
    unsigned short* __restrict__ xbf) {
  __shared__ float sred[256];
  int row = blockIdx.x;                 // b*L + l
  int b = row / Lseq;
  const float* x = hs + (size_t)row * HID;
  float ss = 0.f;
  for (int i = 0; i < HID / 256; ++i) { float v = x[threadIdx.x + i * 256]; ss += v * v; }
  ss = blkSum(ss, sred);
  float rms = rsqrtf(ss * (1.f / HID) + EPSF);
  for (int i = 0; i < HID / 256; ++i) {
    int c = threadIdx.x + i * 256;
    float v = x[c] * rms;
    v = v * (1.f + scale[b * HID + c]) + shift[b * HID + c];
    xbf[(size_t)row * HID + c] = f2bf(v);
  }
}

__global__ void f2bf_kernel(const float* __restrict__ src, unsigned short* __restrict__ dst, int n) {
  int i = blockIdx.x * blockDim.x + threadIdx.x;
  if (i < n) dst[i] = f2bf(src[i]);
}

// W1T[h][f][d] = W1[h][d][f]
__global__ void w1t_kernel(const float* __restrict__ W1, unsigned short* __restrict__ dst) {
  int o = blockIdx.x * blockDim.x + threadIdx.x;
  if (o >= NH * HD * HD) return;
  int h = o / (HD * HD); int r = o % (HD * HD); int f = r / HD; int d = r % HD;
  dst[o] = f2bf(W1[h * HD * HD + d * HD + f]);
}

// C[M,N](f32) = A[M,K](bf16) * Wrow[N,K](bf16)^T (+bias), batched; optional heads-layout out.
// Double-buffered LDS tiles fed by async global->LDS loads.
__global__ __launch_bounds__(256) void gemm_bf16_kernel(
    const unsigned short* __restrict__ A, const unsigned short* __restrict__ W,
    const float* __restrict__ bias, float* __restrict__ C, unsigned short* __restrict__ Cbf,
    int M, int N, int K,
    long long sA, long long sW, int wmod, long long sBias, int bmod,
    long long sC, int out_mode) {
  __shared__ unsigned short As[2][128][32];
  __shared__ unsigned short Bs[2][128][32];
  int tid = threadIdx.x;
  int lane = tid & 31, wave = tid >> 5;
  int wr = wave & 3, wc = wave >> 2;        // 4x2 waves -> 128x128 block tile
  int m0 = blockIdx.y * 128, n0 = blockIdx.x * 128;
  int bz = blockIdx.z;
  const unsigned short* Ab = A + (long long)bz * sA;
  const unsigned short* Wb = W + (long long)(bz % wmod) * sW;

  const v8f vz = {0.f, 0.f, 0.f, 0.f, 0.f, 0.f, 0.f, 0.f};
  v8f acc[2][4];
  for (int i = 0; i < 2; ++i) for (int j = 0; j < 4; ++j) acc[i][j] = vz;

  int lr = lane & 15;
  int kb8 = (lane >> 4) << 3;

  auto issue_tile = [&](int k0, int buf) {
    for (int c = tid; c < 512; c += 256) {
      int row = c >> 2, col8 = (c & 3) << 3;
      async_b128(&As[buf][row][col8], Ab + (long long)(m0 + row) * K + k0 + col8);
      async_b128(&Bs[buf][row][col8], Wb + (long long)(n0 + row) * K + k0 + col8);
    }
  };

  int nk = K / 32;
  issue_tile(0, 0);
  for (int t = 0; t < nk; ++t) {
    int buf = t & 1;
    if (t + 1 < nk) {
      issue_tile((t + 1) * 32, buf ^ 1);
      wait_asynccnt<4>();     // current tile complete; next tile (4 ops/thread) in flight
    } else {
      wait_asynccnt<0>();
    }
    __syncthreads();

    Frag af[2], bfr[4];
#pragma unroll
    for (int q = 0; q < 2; ++q) {
      int row = wr * 32 + q * 16 + lr;
#pragma unroll
      for (int i = 0; i < 4; ++i) {
        af[q].u[i]     = *(const unsigned int*)&As[buf][row][kb8 + 2 * i];
        af[q].u[4 + i] = *(const unsigned int*)&As[buf][row][kb8 + 16 + 2 * i];
      }
    }
#pragma unroll
    for (int q = 0; q < 4; ++q) {
      int row = wc * 64 + q * 16 + lr;
#pragma unroll
      for (int i = 0; i < 4; ++i) {
        bfr[q].u[i]     = *(const unsigned int*)&Bs[buf][row][kb8 + 2 * i];
        bfr[q].u[4 + i] = *(const unsigned int*)&Bs[buf][row][kb8 + 16 + 2 * i];
      }
    }
#pragma unroll
    for (int i = 0; i < 2; ++i)
#pragma unroll
      for (int j = 0; j < 4; ++j)
        acc[i][j] = __builtin_amdgcn_wmma_f32_16x16x32_bf16(
            false, af[i].v, false, bfr[j].v, (short)0, acc[i][j], false, false);
    __syncthreads();
  }

  int rowadd = (lane >> 4) << 3;
#pragma unroll
  for (int i = 0; i < 2; ++i)
#pragma unroll
    for (int j = 0; j < 4; ++j) {
      int mbase = m0 + wr * 32 + i * 16;
      int nbase = n0 + wc * 64 + j * 16;
#pragma unroll
      for (int r = 0; r < 8; ++r) {
        int row = mbase + r + rowadd;
        int col = nbase + lr;
        float v = acc[i][j][r];
        if (bias) v += bias[(long long)(bz % bmod) * sBias + col];
        long long idx;
        if (out_mode == 0) idx = (long long)bz * sC + (long long)row * N + col;
        else {
          int bb = row / Lseq, l = row % Lseq, h = col / HD, d = col % HD;
          idx = (((long long)(bb * NH + h)) * Lseq + l) * HD + d;
        }
        C[idx] = v;
        if (Cbf) Cbf[idx] = f2bf(v);
      }
    }
}

// fused LN-L2 backward per (b,h,l) row -> gradT (bf16, [D,L]) + gradsum (atomic f32)
__global__ __launch_bounds__(128) void lnbwd_kernel(
    const float* __restrict__ Z1, const float* __restrict__ XK, const float* __restrict__ XV,
    const float* __restrict__ tw, const float* __restrict__ tb,
    unsigned short* __restrict__ gradT, float* __restrict__ gradsum) {
  __shared__ float sred[128];
  int l = blockIdx.x;
  int bz = blockIdx.y;                  // b*NH + h
  int h = bz % NH;
  int d = threadIdx.x;
  long long base = ((long long)bz * Lseq + l) * HD + d;
  float x = Z1[base];
  float tgt = XV[base] - XK[base];
  float mu = blkSum(x, sred) * (1.f / HD);
  float xc = x - mu;
  float var = blkSum(xc * xc, sred) * (1.f / HD);
  float stdv = sqrtf(var + EPSF);
  float xh = xc / stdv;
  float gamma = tw[h * HD + d], beta = tb[h * HD + d];
  float g = (gamma * xh + beta - tgt) * gamma;
  float sg = blkSum(g, sred);
  float sgx = blkSum(g * xh, sred);
  float z = ((float)HD * g - sg - xh * sgx) / ((float)HD * stdv);
  gradT[((long long)bz * HD + d) * Lseq + l] = f2bf(z);
  atomicAdd(&gradsum[bz * HD + d], z);
}

// causal dual-form block kernel: Z1_bar = XQ@W1 - eta*(tril(XQ XK^T) @ grad) + b1 - eta*gradsum
// then out_pre = XQ + ln_fwd(Z1_bar)
__global__ __launch_bounds__(128) void attn_kernel(
    const unsigned short* __restrict__ XQbf, const unsigned short* __restrict__ XKbf,
    const unsigned short* __restrict__ gradT, const float* __restrict__ XQf,
    const unsigned short* __restrict__ W1T, const float* __restrict__ b1,
    const float* __restrict__ gradsum, const float* __restrict__ tw, const float* __restrict__ tb,
    float* __restrict__ out_pre) {
  __shared__ __align__(16) unsigned char smem[57344];
  unsigned short* Qs = (unsigned short*)smem;            // [64][128] 16KB
  unsigned short* Ks = (unsigned short*)(smem + 16384);  // [64][128] 16KB
  unsigned short* Gs = (unsigned short*)(smem + 32768);  // [128][64] 16KB (gradT tile [f][m])
  unsigned short* Ss = (unsigned short*)(smem + 49152);  // [64][64]   8KB
  float* Zs   = (float*)(smem + 16384);                  // [64][128] 32KB overlay on Ks+Gs
  float* redm = (float*)(smem + 49152);                  // [64]
  float* redr = redm + 64;                               // [64]

  int tid = threadIdx.x;
  int lane = tid & 31, wave = tid >> 5;
  int rb = blockIdx.x, bz = blockIdx.y;
  int b = bz / NH, h = bz % NH;
  const unsigned short* Q   = XQbf + (long long)bz * Lseq * HD + (long long)rb * 64 * HD;
  const unsigned short* Kp  = XKbf + (long long)bz * Lseq * HD;
  const unsigned short* Gp  = gradT + (long long)bz * HD * Lseq;
  const unsigned short* W1h = W1T + h * HD * HD;

  // warm W1T block into cache (global_prefetch_b8)
  __builtin_prefetch(W1h + (size_t)tid * 128, 0, 3);

  for (int c = tid; c < 1024; c += 128) {
    int row = c >> 4, col8 = (c & 15) << 3;
    async_b128(&Qs[row * 128 + col8], Q + row * HD + col8);
  }

  const v8f vz = {0.f, 0.f, 0.f, 0.f, 0.f, 0.f, 0.f, 0.f};
  v8f accA[8];
  for (int j = 0; j < 8; ++j) accA[j] = vz;

  int ml = wave * 16;                 // each of 4 waves owns 16 output rows
  int lr = lane & 15;
  int kb8 = (lane >> 4) << 3;
  int rowadd = (lane >> 4) << 3;
  wait_asynccnt<0>();
  __syncthreads();

  for (int cb = 0; cb <= rb; ++cb) {
    for (int c = tid; c < 1024; c += 128) {
      int row = c >> 4, col8 = (c & 15) << 3;
      async_b128(&Ks[row * 128 + col8], Kp + (long long)(cb * 64 + row) * HD + col8);
    }
    for (int c = tid; c < 1024; c += 128) {
      int f = c >> 3, col8 = (c & 7) << 3;
      async_b128(&Gs[f * 64 + col8], Gp + (long long)f * Lseq + cb * 64 + col8);
    }
    wait_asynccnt<0>();
    __syncthreads();

    // S(16x64 slice) = XQ . XK^T over d (K=128)
    v8f sac[4];
    for (int j = 0; j < 4; ++j) sac[j] = vz;
#pragma unroll
    for (int kk = 0; kk < 4; ++kk) {
      Frag a; int arow = ml + lr; int kb = kk * 32 + kb8;
#pragma unroll
      for (int i = 0; i < 4; ++i) {
        a.u[i]     = *(const unsigned int*)&Qs[arow * 128 + kb + 2 * i];
        a.u[4 + i] = *(const unsigned int*)&Qs[arow * 128 + kb + 16 + 2 * i];
      }
#pragma unroll
      for (int j = 0; j < 4; ++j) {
        Frag bb; int nrow = j * 16 + lr;
#pragma unroll
        for (int i = 0; i < 4; ++i) {
          bb.u[i]     = *(const unsigned int*)&Ks[nrow * 128 + kb + 2 * i];
          bb.u[4 + i] = *(const unsigned int*)&Ks[nrow * 128 + kb + 16 + 2 * i];
        }
        sac[j] = __builtin_amdgcn_wmma_f32_16x16x32_bf16(
            false, a.v, false, bb.v, (short)0, sac[j], false, false);
      }
    }
    // causal mask (diag block only) + bf16 into LDS
#pragma unroll
    for (int j = 0; j < 4; ++j)
#pragma unroll
      for (int r = 0; r < 8; ++r) {
        int row = ml + r + rowadd;
        int col = j * 16 + lr;
        float v = sac[j][r];
        if (cb == rb && col > row) v = 0.f;
        Ss[row * 64 + col] = f2bf(v);
      }
    __syncthreads();

    // accA += S @ grad  (K = m, 64)
#pragma unroll
    for (int kk = 0; kk < 2; ++kk) {
      Frag a; int arow = ml + lr; int kb = kk * 32 + kb8;
#pragma unroll
      for (int i = 0; i < 4; ++i) {
        a.u[i]     = *(const unsigned int*)&Ss[arow * 64 + kb + 2 * i];
        a.u[4 + i] = *(const unsigned int*)&Ss[arow * 64 + kb + 16 + 2 * i];
      }
#pragma unroll
      for (int j = 0; j < 8; ++j) {
        Frag bb; int nrow = j * 16 + lr;      // f
#pragma unroll
        for (int i = 0; i < 4; ++i) {
          bb.u[i]     = *(const unsigned int*)&Gs[nrow * 64 + kb + 2 * i];
          bb.u[4 + i] = *(const unsigned int*)&Gs[nrow * 64 + kb + 16 + 2 * i];
        }
        accA[j] = __builtin_amdgcn_wmma_f32_16x16x32_bf16(
            false, a.v, false, bb.v, (short)0, accA[j], false, false);
      }
    }
    __syncthreads();
  }

  // qw = XQ @ W1 (B frags straight from global W1T; tiny, L2-resident)
  v8f qw[8];
  for (int j = 0; j < 8; ++j) qw[j] = vz;
#pragma unroll
  for (int kk = 0; kk < 4; ++kk) {
    Frag a; int arow = ml + lr; int kb = kk * 32 + kb8;
#pragma unroll
    for (int i = 0; i < 4; ++i) {
      a.u[i]     = *(const unsigned int*)&Qs[arow * 128 + kb + 2 * i];
      a.u[4 + i] = *(const unsigned int*)&Qs[arow * 128 + kb + 16 + 2 * i];
    }
#pragma unroll
    for (int j = 0; j < 8; ++j) {
      Frag bb; int nrow = j * 16 + lr;
#pragma unroll
      for (int i = 0; i < 4; ++i) {
        bb.u[i]     = *(const unsigned int*)(W1h + nrow * HD + kb + 2 * i);
        bb.u[4 + i] = *(const unsigned int*)(W1h + nrow * HD + kb + 16 + 2 * i);
      }
      qw[j] = __builtin_amdgcn_wmma_f32_16x16x32_bf16(
          false, a.v, false, bb.v, (short)0, qw[j], false, false);
    }
  }

  // combine Z1_bar into Zs (f32, overlays Ks/Gs)
  const float eta = 1.f / (float)HD;
#pragma unroll
  for (int j = 0; j < 8; ++j)
#pragma unroll
    for (int r = 0; r < 8; ++r) {
      int row = ml + r + rowadd;
      int col = j * 16 + lr;
      float v = qw[j][r] - eta * accA[j][r] + b1[h * HD + col] - eta * gradsum[bz * HD + col];
      Zs[row * 128 + col] = v;
    }
  __syncthreads();

  if (tid < 64) {
    float mu = 0.f;
    for (int c = 0; c < HD; ++c) mu += Zs[tid * 128 + c];
    mu *= (1.f / HD);
    float var = 0.f;
    for (int c = 0; c < HD; ++c) { float d0 = Zs[tid * 128 + c] - mu; var += d0 * d0; }
    var *= (1.f / HD);
    redm[tid] = mu;
    redr[tid] = rsqrtf(var + EPSF);
  }
  __syncthreads();

  const float* Qf = XQf + (long long)bz * Lseq * HD;
  for (int e = tid; e < 64 * 128; e += 128) {
    int r = e >> 7, c = e & 127;
    float v = tw[h * HD + c] * ((Zs[r * 128 + c] - redm[r]) * redr[r]) + tb[h * HD + c];
    int l = rb * 64 + r;
    out_pre[(((long long)(b * Lseq + l)) * NH + h) * HD + c] =
        Qf[(long long)(rb * 64 + r) * HD + c] + v;
  }
}

// post-LayerNorm over hidden (2048) with affine -> bf16 for the Wo GEMM
__global__ __launch_bounds__(256) void postln_kernel(const float* __restrict__ x,
    const float* __restrict__ pw, const float* __restrict__ pb,
    unsigned short* __restrict__ nrm) {
  __shared__ float sred[256];
  long long row = blockIdx.x;
  const float* xr = x + row * HID;
  float sm = 0.f;
  for (int i = 0; i < HID / 256; ++i) sm += xr[threadIdx.x + i * 256];
  float mu = blkSum(sm, sred) * (1.f / HID);
  float sv = 0.f;
  for (int i = 0; i < HID / 256; ++i) { float d0 = xr[threadIdx.x + i * 256] - mu; sv += d0 * d0; }
  float rstd = rsqrtf(blkSum(sv, sred) * (1.f / HID) + EPSF);
  for (int i = 0; i < HID / 256; ++i) {
    int c = threadIdx.x + i * 256;
    nrm[row * HID + c] = f2bf((xr[c] - mu) * rstd * pw[c] + pb[c]);
  }
}

extern "C" void kernel_launch(void* const* d_in, const int* in_sizes, int n_in,
                              void* d_out, int out_size, void* d_ws, size_t ws_size,
                              hipStream_t stream) {
  (void)in_sizes; (void)n_in; (void)out_size; (void)ws_size;
  const float* hs   = (const float*)d_in[0];
  const float* cond = (const float*)d_in[1];
  const float* Wq   = (const float*)d_in[2];
  const float* Wk   = (const float*)d_in[3];
  const float* Wv   = (const float*)d_in[4];
  const float* Wo   = (const float*)d_in[5];
  const float* dw   = (const float*)d_in[6];
  const float* db   = (const float*)d_in[7];
  const float* tw   = (const float*)d_in[8];
  const float* tb   = (const float*)d_in[9];
  const float* pw   = (const float*)d_in[10];
  const float* pb   = (const float*)d_in[11];
  const float* W1   = (const float*)d_in[12];
  const float* b1   = (const float*)d_in[13];
  float* out = (float*)d_out;
  float* gate_out = out + (long long)Bsz * Lseq * HID;

  char* w = (char*)d_ws;
  auto alloc = [&](size_t bytes) { char* p = w; w += (bytes + 255) & ~(size_t)255; return p; };
  float* scale = (float*)alloc((size_t)Bsz * HID * 4);
  float* shift = (float*)alloc((size_t)Bsz * HID * 4);
  unsigned short* xbf = (unsigned short*)alloc((size_t)Bsz * Lseq * HID * 2);
  unsigned short* Wqb = (unsigned short*)alloc((size_t)HID * HID * 2);
  unsigned short* Wkb = (unsigned short*)alloc((size_t)HID * HID * 2);
  unsigned short* Wvb = (unsigned short*)alloc((size_t)HID * HID * 2);
  unsigned short* Wob = (unsigned short*)alloc((size_t)HID * HID * 2);
  unsigned short* W1T = (unsigned short*)alloc((size_t)NH * HD * HD * 2);
  float* XQ = (float*)alloc((size_t)Bsz * NH * Lseq * HD * 4);
  float* XK = (float*)alloc((size_t)Bsz * NH * Lseq * HD * 4);
  float* XV = (float*)alloc((size_t)Bsz * NH * Lseq * HD * 4);
  unsigned short* XQb = (unsigned short*)alloc((size_t)Bsz * NH * Lseq * HD * 2);
  unsigned short* XKb = (unsigned short*)alloc((size_t)Bsz * NH * Lseq * HD * 2);
  float* Z1 = (float*)alloc((size_t)Bsz * NH * Lseq * HD * 4);
  unsigned short* gradT = (unsigned short*)alloc((size_t)Bsz * NH * HD * Lseq * 2);
  float* gradsum = (float*)alloc((size_t)Bsz * NH * HD * 4);
  float* out_pre = (float*)alloc((size_t)Bsz * Lseq * HID * 4);
  unsigned short* nrm = (unsigned short*)alloc((size_t)Bsz * Lseq * HID * 2);

  (void)hipMemsetAsync(gradsum, 0, (size_t)Bsz * NH * HD * 4, stream);

  mod_kernel<<<(Bsz * 3 * HID + 255) / 256, 256, 0, stream>>>(cond, dw, db, scale, shift, gate_out);
  adanorm_kernel<<<Bsz * Lseq, 256, 0, stream>>>(hs, scale, shift, xbf);

  int nW = HID * HID;
  f2bf_kernel<<<(nW + 255) / 256, 256, 0, stream>>>(Wq, Wqb, nW);
  f2bf_kernel<<<(nW + 255) / 256, 256, 0, stream>>>(Wk, Wkb, nW);
  f2bf_kernel<<<(nW + 255) / 256, 256, 0, stream>>>(Wv, Wvb, nW);
  f2bf_kernel<<<(nW + 255) / 256, 256, 0, stream>>>(Wo, Wob, nW);
  w1t_kernel<<<(NH * HD * HD + 255) / 256, 256, 0, stream>>>(W1, W1T);

  dim3 gQKV(HID / 128, (Bsz * Lseq) / 128, 1);
  gemm_bf16_kernel<<<gQKV, 256, 0, stream>>>(xbf, Wqb, nullptr, XQ, XQb,
      Bsz * Lseq, HID, HID, 0LL, 0LL, 1, 0LL, 1, 0LL, 1);
  gemm_bf16_kernel<<<gQKV, 256, 0, stream>>>(xbf, Wkb, nullptr, XK, XKb,
      Bsz * Lseq, HID, HID, 0LL, 0LL, 1, 0LL, 1, 0LL, 1);
  gemm_bf16_kernel<<<gQKV, 256, 0, stream>>>(xbf, Wvb, nullptr, XV, nullptr,
      Bsz * Lseq, HID, HID, 0LL, 0LL, 1, 0LL, 1, 0LL, 1);

  dim3 gZ1(HD / 128, Lseq / 128, Bsz * NH);
  gemm_bf16_kernel<<<gZ1, 256, 0, stream>>>(XKb, W1T, b1, Z1, nullptr,
      Lseq, HD, HD, (long long)Lseq * HD, (long long)HD * HD, NH,
      (long long)HD, NH, (long long)Lseq * HD, 0);

  dim3 gBwd(Lseq, Bsz * NH);
  lnbwd_kernel<<<gBwd, 128, 0, stream>>>(Z1, XK, XV, tw, tb, gradT, gradsum);

  dim3 gAtt(Lseq / 64, Bsz * NH);
  attn_kernel<<<gAtt, 128, 0, stream>>>(XQb, XKb, gradT, XQ, W1T, b1, gradsum, tw, tb, out_pre);

  postln_kernel<<<Bsz * Lseq, 256, 0, stream>>>(out_pre, pw, pb, nrm);

  dim3 gO(HID / 128, (Bsz * Lseq) / 128, 1);
  gemm_bf16_kernel<<<gO, 256, 0, stream>>>(nrm, Wob, nullptr, out, nullptr,
      Bsz * Lseq, HID, HID, 0LL, 0LL, 1, 0LL, 1, (long long)Bsz * Lseq * HID, 0);
}
